// UzmanHavuzu_30030411334253
// MI455X (gfx1250) — matrix operand order repaired and checked by reading, not compiled, not run on hardware
//
#include <hip/hip_runtime.h>
#include <math.h>

// Problem constants (from reference): N=16384, D=1024, H=2048, E=16, K=2
#define NTOK 16384
#define DDIM 1024
#define HDIM 2048
#define NEXP 16
#define TOPK 2
#define NPAIR (NTOK * TOPK) // 32768 (token, slot) pairs

// Tile config: 128x256 block tile, 256 threads = 8 waves arranged 2(M) x 4(N),
// each wave owns a 64x64 tile = 4x4 WMMA 16x16x32 tiles (16 wmma / 8 ds_load_b128 for A + 64 16-bit loads for B).
#define TM 128
#define TN 256
#define KB 32
#define APAD 40   // ushorts per A LDS row (80B stride, mult of 16B -> aligned b128)
#define BPAD 264  // ushorts per B LDS row (528B stride, mult of 16B, low conflict)

#if __has_builtin(__builtin_amdgcn_global_load_async_to_lds_b128)
#define HAVE_ASYNC_LDS 1
#else
#define HAVE_ASYNC_LDS 0
#endif

typedef __attribute__((ext_vector_type(16))) __bf16 v16bf;
typedef __attribute__((ext_vector_type(8)))  float v8f;
typedef __attribute__((ext_vector_type(4)))  int   v4i;
typedef __attribute__((address_space(1))) v4i* gv4i_ptr;
typedef __attribute__((address_space(3))) v4i* lv4i_ptr;

union Frag { v16bf v; uint4 q[2]; };

__device__ __forceinline__ unsigned short f2bf(float f) {
  union { float f; unsigned int u; } c; c.f = f;
  unsigned int u = c.u;
  return (unsigned short)((u + 0x7FFFu + ((u >> 16) & 1u)) >> 16); // RNE
}

__device__ __forceinline__ void async_wait0() {
#if HAVE_ASYNC_LDS
#if __has_builtin(__builtin_amdgcn_s_wait_asynccnt)
  __builtin_amdgcn_s_wait_asynccnt(0);
#else
  asm volatile("s_wait_asynccnt 0" ::: "memory");
#endif
#endif
}

// ---------------------------------------------------------------------------
// Kernel 1: routing. Build per-expert contiguous pair lists.
// ---------------------------------------------------------------------------
__global__ __launch_bounds__(256) void route_kernel(
    const int* __restrict__ idx, const float* __restrict__ wts,
    int* __restrict__ offs, int* __restrict__ tok, float* __restrict__ wgt) {
  __shared__ int cnt[NEXP];
  __shared__ int base[NEXP + 1];
  __shared__ int fill[NEXP];
  const int t = threadIdx.x;
  if (t < NEXP) { cnt[t] = 0; fill[t] = 0; }
  __syncthreads();
  for (int i = t; i < NPAIR; i += 256) atomicAdd(&cnt[idx[i]], 1);
  __syncthreads();
  if (t == 0) {
    int s = 0;
    for (int e = 0; e < NEXP; ++e) { base[e] = s; s += cnt[e]; }
    base[NEXP] = s;
  }
  __syncthreads();
  if (t < NEXP + 1) offs[t] = base[t];
  for (int i = t; i < NPAIR; i += 256) {
    const int e = idx[i];
    const int p = base[e] + atomicAdd(&fill[e], 1);
    tok[p] = i >> 1;   // K == 2
    wgt[p] = wts[i];
  }
}

// ---------------------------------------------------------------------------
// Kernel 2: h = gelu_exact(x_gather @ W1[e] + b1[e]), bf16 WMMA, double-buffered.
// grid = (H/TN, maxRowTiles, E)
// ---------------------------------------------------------------------------
__global__ __launch_bounds__(256) void gemm1_gelu_kernel(
    const float* __restrict__ x, const float* __restrict__ W1,
    const float* __restrict__ b1, const int* __restrict__ offs,
    const int* __restrict__ tok, unsigned short* __restrict__ h) {
  const int e = blockIdx.z;
  const int row_lo = offs[e], row_hi = offs[e + 1];
  const int row0 = row_lo + blockIdx.y * TM;
  if (row0 >= row_hi) return;
  const int col0 = blockIdx.x * TN;
  int rows = row_hi - row0; if (rows > TM) rows = TM;

  __shared__ __align__(16) unsigned short As[2][TM * APAD];
  __shared__ __align__(16) unsigned short Bs[2][KB * BPAD];
  __shared__ int toks[TM];

  const int tid = threadIdx.x;
  if (tid < TM) {
    const int r = tid < rows ? tid : rows - 1;
    toks[tid] = tok[row0 + r];
  }
  __syncthreads();

  const int lane = tid & 31, wave = tid >> 5;
  const int wm = wave >> 2, wn = wave & 3;   // 2 x 4 waves, 64x64 per wave

  auto stageA = [&](unsigned short* dst, int kk) {
    #pragma unroll
    for (int i = 0; i < 4; ++i) {
      const int c = tid + i * 256;       // 1024 float4 chunks (128 rows x 8)
      const int r = c >> 3;
      const int cc = (c & 7) * 4;
      const float4 v = *(const float4*)(x + (size_t)toks[r] * DDIM + kk + cc);
      unsigned short* d = dst + r * APAD + cc;
      d[0] = f2bf(v.x); d[1] = f2bf(v.y); d[2] = f2bf(v.z); d[3] = f2bf(v.w);
    }
  };
  auto stageB = [&](unsigned short* dst, int kk) {
    #pragma unroll
    for (int i = 0; i < 8; ++i) {
      const int c = tid + i * 256;       // 2048 float4 chunks (32 rows x 64)
      const int r = c >> 6;
      const int cc = (c & 63) * 4;
      const float4 v = *(const float4*)(W1 + ((size_t)e * DDIM + (kk + r)) * HDIM + col0 + cc);
      unsigned short* d = dst + r * BPAD + cc;
      d[0] = f2bf(v.x); d[1] = f2bf(v.y); d[2] = f2bf(v.z); d[3] = f2bf(v.w);
    }
  };

  v8f acc[4][4];
  #pragma unroll
  for (int mi = 0; mi < 4; ++mi)
    #pragma unroll
    for (int ni = 0; ni < 4; ++ni)
      acc[mi][ni] = (v8f){0.f, 0.f, 0.f, 0.f, 0.f, 0.f, 0.f, 0.f};

  stageA(As[0], 0);
  stageB(Bs[0], 0);
  __syncthreads();

  int buf = 0;
  for (int k0 = 0; k0 < DDIM; k0 += KB) {
    // Prefetch k0+2*KB toward L2/L0 (one 128B line per thread covers the panels)
    if (k0 + 2 * KB < DDIM) {
      __builtin_prefetch(W1 + ((size_t)e * DDIM + (k0 + 2 * KB + (tid >> 3))) * HDIM +
                             col0 + ((tid & 7) << 5), 0, 3);
      if (tid < TM)
        __builtin_prefetch(x + (size_t)toks[tid] * DDIM + k0 + 2 * KB, 0, 3);
    }
    // A frag (16x32 bf16): lane<16 row holds K {0..7,16..23}; lane>=16 K {8..15,24..31}
    Frag a[4], b[4];
    const int am = (lane & 15);
    const int kb = (lane < 16) ? 0 : 8;
    #pragma unroll
    for (int mi = 0; mi < 4; ++mi) {
      const unsigned short* p = &As[buf][(wm * 64 + mi * 16 + am) * APAD + kb];
      a[mi].q[0] = *(const uint4*)p;
      a[mi].q[1] = *(const uint4*)(p + 16);
    }
    // B frag (32x16 bf16): lane holds column n=lane%16; lane<16 K {0..7,16..23}; lane>=16 K {8..15,24..31}
    #pragma unroll
    for (int ni = 0; ni < 4; ++ni) {
      const unsigned short* p = &Bs[buf][kb * BPAD + wn * 64 + ni * 16 + am];
      #pragma unroll
      for (int s = 0; s < 8; ++s) {
        ((unsigned short*)&b[ni])[s] = p[s * BPAD];
        ((unsigned short*)&b[ni])[8 + s] = p[(16 + s) * BPAD];
      }
    }
    #pragma unroll
    for (int mi = 0; mi < 4; ++mi)
      #pragma unroll
      for (int ni = 0; ni < 4; ++ni)
        acc[mi][ni] = __builtin_amdgcn_wmma_f32_16x16x32_bf16(
            false, a[mi].v, false, b[ni].v, (short)0, acc[mi][ni], false, false);
    // Stage next K-chunk into the other buffer (overlaps with this iter's WMMAs)
    if (k0 + KB < DDIM) {
      stageA(As[buf ^ 1], k0 + KB);
      stageB(Bs[buf ^ 1], k0 + KB);
    }
    __syncthreads();
    buf ^= 1;
  }

  // Epilogue: +b1, exact-erf GELU, bf16 store to h
  const int nbase = col0 + wn * 64;
  #pragma unroll
  for (int ni = 0; ni < 4; ++ni) {
    const int col = nbase + ni * 16 + (lane & 15);
    const float bias = b1[(size_t)e * HDIM + col];
    #pragma unroll
    for (int mi = 0; mi < 4; ++mi) {
      #pragma unroll
      for (int v = 0; v < 8; ++v) {
        const int lm = wm * 64 + mi * 16 + v + ((lane >> 4) << 3);
        if (lm < rows) {
          const float t = acc[mi][ni][v] + bias;
          const float g = 0.5f * t * (1.0f + erff(t * 0.70710678118654752f));
          h[(size_t)(row0 + lm) * HDIM + col] = f2bf(g);
        }
      }
    }
  }
}

// ---------------------------------------------------------------------------
// Kernel 3: y = h @ W2[e] + b2[e]; out[token] += gate * y (atomic, 2 adds/elem)
// A-tiles (bf16 h) staged via async global->LDS when available. grid = (D/TN, maxRowTiles, E)
// ---------------------------------------------------------------------------
__global__ __launch_bounds__(256) void gemm2_scatter_kernel(
    const unsigned short* __restrict__ h, const float* __restrict__ W2,
    const float* __restrict__ b2, const int* __restrict__ offs,
    const int* __restrict__ tok, const float* __restrict__ wgt,
    float* __restrict__ out) {
  const int e = blockIdx.z;
  const int row_lo = offs[e], row_hi = offs[e + 1];
  const int row0 = row_lo + blockIdx.y * TM;
  if (row0 >= row_hi) return;
  const int col0 = blockIdx.x * TN;
  int rows = row_hi - row0; if (rows > TM) rows = TM;

  __shared__ __align__(16) unsigned short As[2][TM * APAD];
  __shared__ __align__(16) unsigned short Bs[2][KB * BPAD];
  __shared__ int toks[TM];
  __shared__ float wg[TM];

  const int tid = threadIdx.x;
  if (tid < TM) {
    const int r = tid < rows ? tid : rows - 1;
    toks[tid] = tok[row0 + r];
    wg[tid] = wgt[row0 + r];
  }
  __syncthreads();

  const int lane = tid & 31, wave = tid >> 5;
  const int wm = wave >> 2, wn = wave & 3;

  auto stageA = [&](unsigned short* dst, int kk) {
    #pragma unroll
    for (int i = 0; i < 2; ++i) {
      const int c = tid + i * 256;       // 512 16B chunks (128 rows x 4)
      const int r = c >> 2;
      const int cc = (c & 3) * 8;
      const int rr = r < rows ? r : rows - 1;
      const unsigned short* src = h + (size_t)(row0 + rr) * HDIM + kk + cc;
#if HAVE_ASYNC_LDS
      __builtin_amdgcn_global_load_async_to_lds_b128(
          (gv4i_ptr)src, (lv4i_ptr)(dst + r * APAD + cc), 0, 0);
#else
      *(uint4*)(dst + r * APAD + cc) = *(const uint4*)src;
#endif
    }
  };
  auto stageB = [&](unsigned short* dst, int kk) {
    #pragma unroll
    for (int i = 0; i < 8; ++i) {
      const int c = tid + i * 256;       // 2048 float4 chunks (32 rows x 64)
      const int r = c >> 6;
      const int cc = (c & 63) * 4;
      const float4 v = *(const float4*)(W2 + ((size_t)e * HDIM + (kk + r)) * DDIM + col0 + cc);
      unsigned short* d = dst + r * BPAD + cc;
      d[0] = f2bf(v.x); d[1] = f2bf(v.y); d[2] = f2bf(v.z); d[3] = f2bf(v.w);
    }
  };

  v8f acc[4][4];
  #pragma unroll
  for (int mi = 0; mi < 4; ++mi)
    #pragma unroll
    for (int ni = 0; ni < 4; ++ni)
      acc[mi][ni] = (v8f){0.f, 0.f, 0.f, 0.f, 0.f, 0.f, 0.f, 0.f};

  stageA(As[0], 0);
  stageB(Bs[0], 0);
  async_wait0();
  __syncthreads();

  int buf = 0;
  for (int k0 = 0; k0 < HDIM; k0 += KB) {
    if (k0 + 2 * KB < HDIM) {
      __builtin_prefetch(W2 + ((size_t)e * HDIM + (k0 + 2 * KB + (tid >> 3))) * DDIM +
                             col0 + ((tid & 7) << 5), 0, 3);
      if (tid < TM)
        __builtin_prefetch(h + (size_t)(row0 + (tid < rows ? tid : rows - 1)) * HDIM +
                               k0 + 2 * KB, 0, 3);
    }
    Frag a[4], b[4];
    const int am = (lane & 15);
    const int kb = (lane < 16) ? 0 : 8;
    #pragma unroll
    for (int mi = 0; mi < 4; ++mi) {
      const unsigned short* p = &As[buf][(wm * 64 + mi * 16 + am) * APAD + kb];
      a[mi].q[0] = *(const uint4*)p;
      a[mi].q[1] = *(const uint4*)(p + 16);
    }
    #pragma unroll
    for (int ni = 0; ni < 4; ++ni) {
      const unsigned short* p = &Bs[buf][kb * BPAD + wn * 64 + ni * 16 + am];
      #pragma unroll
      for (int s = 0; s < 8; ++s) {
        ((unsigned short*)&b[ni])[s] = p[s * BPAD];
        ((unsigned short*)&b[ni])[8 + s] = p[(16 + s) * BPAD];
      }
    }
    #pragma unroll
    for (int mi = 0; mi < 4; ++mi)
      #pragma unroll
      for (int ni = 0; ni < 4; ++ni)
        acc[mi][ni] = __builtin_amdgcn_wmma_f32_16x16x32_bf16(
            false, a[mi].v, false, b[ni].v, (short)0, acc[mi][ni], false, false);
    if (k0 + KB < HDIM) {
      stageA(As[buf ^ 1], k0 + KB);
      stageB(Bs[buf ^ 1], k0 + KB);
    }
    async_wait0();
    __syncthreads();
    buf ^= 1;
  }

  // Epilogue: +b2, scale by gate, scatter-add to out (exactly 2 adds per out elem)
  const int nbase = col0 + wn * 64;
  #pragma unroll
  for (int ni = 0; ni < 4; ++ni) {
    const int col = nbase + ni * 16 + (lane & 15);
    const float bias = b2[(size_t)e * DDIM + col];
    #pragma unroll
    for (int mi = 0; mi < 4; ++mi) {
      #pragma unroll
      for (int v = 0; v < 8; ++v) {
        const int lm = wm * 64 + mi * 16 + v + ((lane >> 4) << 3);
        if (lm < rows) {
          const float val = (acc[mi][ni][v] + bias) * wg[lm];
          atomicAdd(&out[(size_t)toks[lm] * DDIM + col], val);
        }
      }
    }
  }
}

// ---------------------------------------------------------------------------
extern "C" void kernel_launch(void* const* d_in, const int* in_sizes, int n_in,
                              void* d_out, int out_size, void* d_ws, size_t ws_size,
                              hipStream_t stream) {
  const float* x   = (const float*)d_in[0];
  const int*   idx = (const int*)  d_in[1];
  const float* wts = (const float*)d_in[2];
  const float* W1  = (const float*)d_in[3];
  const float* b1  = (const float*)d_in[4];
  const float* W2  = (const float*)d_in[5];
  const float* b2  = (const float*)d_in[6];
  float* out = (float*)d_out;

  // Workspace layout (16B-aligned):
  //   [0,128)              : offs[E+1]
  //   [128, +4*NPAIR)      : tok
  //   [.., +4*NPAIR)       : wgt
  //   [.., +2*NPAIR*HDIM)  : h (bf16, 128 MB)
  char* ws = (char*)d_ws;
  int*   offs = (int*)ws;
  int*   tok  = (int*)(ws + 128);
  float* wgt  = (float*)(ws + 128 + (size_t)NPAIR * 4);
  unsigned short* h = (unsigned short*)(ws + 128 + (size_t)NPAIR * 8);

  (void)hipMemsetAsync(d_out, 0, (size_t)NTOK * DDIM * sizeof(float), stream);

  route_kernel<<<1, 256, 0, stream>>>(idx, wts, offs, tok, wgt);

  const int max_row_tiles = (NPAIR + TM - 1) / TM; // worst case: one expert owns all
  dim3 g1(HDIM / TN, max_row_tiles, NEXP);
  gemm1_gelu_kernel<<<g1, 256, 0, stream>>>(x, W1, b1, offs, tok, h);

  dim3 g2(DDIM / TN, max_row_tiles, NEXP);
  gemm2_scatter_kernel<<<g2, 256, 0, stream>>>(h, W2, b2, offs, tok, wgt, out);
}
